// LtiRnn_17231408791956
// MI455X (gfx1250) — matrix operand order repaired and checked by reading, not compiled, not run on hardware
//
#include <hip/hip_runtime.h>

// LTI-RNN scan for MI455X (gfx1250, wave32, WMMA 16x16x32 bf16, f32 accum).
// Y_inv and diag(1/lambda) are folded into the weights once (prep kernels),
// so each recurrent step is 13 WMMAs + a branch-free fast tanh.
// All WMMAs execute with full EXEC (no predication) per ISA 7.12 restrictions.

typedef __attribute__((ext_vector_type(16))) __bf16 v16bf;
typedef __attribute__((ext_vector_type(8)))  float  v8f;

#define NXD 64
#define NUD 32
#define NYD 32
#define NWD 64
#define BBATCH 256
#define TSTEPS 2048

// --- d_ws layout ---
// [0, 16384)            : Yinv (64x64 f32)
// then bf16 effective weights (element offsets within W):
//   Wz_x [64][64] @ 0      :  C2[j][k]*tinv[j]
//   Wz_u [32][64] @ 4096   :  D21[j][k]*tinv[j]
//   Wx_x [64][64] @ 6144   :  (A^T  Yinv)[k][j]
//   Wx_u [32][64] @ 10240  :  (B1^T Yinv)[k][j]
//   Wx_w [64][64] @ 12288  :  (B2^T Yinv)[k][j]
//   Wy_x [64][32] @ 16384  :  C1[j][k]
//   Wy_u [32][32] @ 18432  :  D11[j][k]
//   Wy_w [64][32] @ 19456  :  D12[j][k]

// ---------------- prep 1: Gauss-Jordan inverse of Y (64x64) ----------------
__global__ __launch_bounds__(128) void prep_inverse(const float* __restrict__ Y,
                                                    float* __restrict__ Yinv) {
  __shared__ float M[64][129];   // augmented [Y | I], padded stride
  __shared__ float fac[64];
  const int tid = threadIdx.x;   // 128 threads, thread tid owns column tid
  for (int idx = tid; idx < 64 * 128; idx += 128) {
    int r = idx >> 7, c = idx & 127;
    M[r][c] = (c < 64) ? Y[r * 64 + c] : ((c - 64) == r ? 1.0f : 0.0f);
  }
  __syncthreads();
  for (int p = 0; p < 64; ++p) {
    float pivinv = 1.0f / M[p][p];
    __syncthreads();
    M[p][tid] *= pivinv;                 // normalize pivot row
    __syncthreads();
    if (tid < 64) fac[tid] = M[tid][p];  // snapshot pivot column
    __syncthreads();
    for (int r = 0; r < 64; ++r)
      if (r != p) M[r][tid] -= fac[r] * M[p][tid];
    __syncthreads();
  }
  for (int idx = tid; idx < 64 * 64; idx += 128)
    Yinv[idx] = M[idx >> 6][64 + (idx & 63)];
}

// ---------------- prep 2: build effective bf16 weights ----------------
__global__ __launch_bounds__(256) void prep_weights(
    const float* __restrict__ lambdas, const float* __restrict__ A,
    const float* __restrict__ B1, const float* __restrict__ B2,
    const float* __restrict__ C1, const float* __restrict__ D11,
    const float* __restrict__ D12, const float* __restrict__ C2,
    const float* __restrict__ D21, const float* __restrict__ Yinv,
    __bf16* __restrict__ W) {
  const int tid = threadIdx.x;
  for (int idx = tid; idx < NXD * NWD; idx += 256) {           // Wz_x
    int k = idx / NWD, j = idx % NWD;
    W[idx] = (__bf16)(C2[j * NXD + k] * (1.0f / lambdas[j]));
  }
  for (int idx = tid; idx < NUD * NWD; idx += 256) {           // Wz_u
    int k = idx / NWD, j = idx % NWD;
    W[4096 + idx] = (__bf16)(D21[j * NUD + k] * (1.0f / lambdas[j]));
  }
  for (int idx = tid; idx < NXD * NXD; idx += 256) {           // Wx_x
    int k = idx / NXD, j = idx % NXD;
    float s = 0.f;
    for (int m = 0; m < NXD; ++m) s += A[m * NXD + k] * Yinv[m * NXD + j];
    W[6144 + idx] = (__bf16)s;
  }
  for (int idx = tid; idx < NUD * NXD; idx += 256) {           // Wx_u
    int k = idx / NXD, j = idx % NXD;
    float s = 0.f;
    for (int m = 0; m < NXD; ++m) s += B1[m * NUD + k] * Yinv[m * NXD + j];
    W[10240 + idx] = (__bf16)s;
  }
  for (int idx = tid; idx < NWD * NXD; idx += 256) {           // Wx_w
    int k = idx / NXD, j = idx % NXD;
    float s = 0.f;
    for (int m = 0; m < NXD; ++m) s += B2[m * NWD + k] * Yinv[m * NXD + j];
    W[12288 + idx] = (__bf16)s;
  }
  for (int idx = tid; idx < NXD * NYD; idx += 256) {           // Wy_x
    int k = idx / NYD, j = idx % NYD;
    W[16384 + idx] = (__bf16)C1[j * NXD + k];
  }
  for (int idx = tid; idx < NUD * NYD; idx += 256) {           // Wy_u
    int k = idx / NYD, j = idx % NYD;
    W[18432 + idx] = (__bf16)D11[j * NUD + k];
  }
  for (int idx = tid; idx < NWD * NYD; idx += 256) {           // Wy_w
    int k = idx / NYD, j = idx % NYD;
    W[19456 + idx] = (__bf16)D12[j * NWD + k];
  }
}

// ---------------- WMMA fragment loaders (wave32, 16x16x32 bf16) ----------------
// A-matrix 16x32 bf16: lane L holds row L&15; lanes 0-15 cover K {0..7,16..23},
// lanes 16-31 cover K {8..15,24..31} (per ISA 7.12.2 table).
__device__ inline v16bf load_a_lds(const __bf16* Xb, int kbase, int lane) {
  int row = lane & 15;
  int kh = (lane >> 4) * 8;
  const __bf16* p = Xb + row * 64 + kbase;
  v16bf a;
#pragma unroll
  for (int i = 0; i < 4; ++i) {
    a[2 * i]     = p[kh + 2 * i];
    a[2 * i + 1] = p[kh + 2 * i + 1];
    a[8 + 2 * i]     = p[16 + kh + 2 * i];
    a[8 + 2 * i + 1] = p[16 + kh + 2 * i + 1];
  }
  return a;
}

__device__ inline v16bf load_a_global_f32(const float* base, long rowStride, int lane) {
  int row = lane & 15;
  int kh = (lane >> 4) * 8;
  const float* p = base + (long)row * rowStride;
  v16bf a;
#pragma unroll
  for (int i = 0; i < 4; ++i) {
    a[2 * i]     = (__bf16)p[kh + 2 * i];
    a[2 * i + 1] = (__bf16)p[kh + 2 * i + 1];
    a[8 + 2 * i]     = (__bf16)p[16 + kh + 2 * i];
    a[8 + 2 * i + 1] = (__bf16)p[16 + kh + 2 * i + 1];
  }
  return a;
}

// B-matrix 32x16 bf16 from row-major [K][ncols]: lane L holds col L&15,
// VGPR v half h -> K = 8*(L>>4) + 2v + h (+8 for v>=4), matching the A order.
__device__ inline v16bf load_b_frag(const __bf16* Wm, int ncols, int colBase, int lane) {
  int col = colBase + (lane & 15);
  int kb = (lane >> 4) * 8;
  v16bf b;
#pragma unroll
  for (int v = 0; v < 8; ++v) {
    int k = kb + 2 * v + (v >= 4 ? 8 : 0);
    b[2 * v]     = Wm[k * ncols + col];
    b[2 * v + 1] = Wm[(k + 1) * ncols + col];
  }
  return b;
}

#define WMMA_BF16(A, Bv, C) \
  __builtin_amdgcn_wmma_f32_16x16x32_bf16(false, (A), false, (Bv), (short)0, (C), false, false)

// Branch-free tanh: 1 - 2*rcp(exp2(2*log2e*x) + 1).
// Saturates correctly: exp->inf => +1 ; exp->0 => -1. ~5 ops, no EXEC branches.
__device__ inline float fast_tanh(float x) {
  float e = __builtin_amdgcn_exp2f(2.885390081777927f * x);  // exp(2x)
  float r = __builtin_amdgcn_rcpf(e + 1.0f);
  return __builtin_fmaf(-2.0f, r, 1.0f);
}

// ---------------- main recurrent kernel ----------------
// Grid: 16 WGs (16 batch rows each). Block: 128 threads = 4 waves.
// Wave n owns N-tile n (cols 16n..16n+15) of z (NW=64) and x_next (NX=64).
// All 4 waves compute a y tile (waves 2,3 duplicate waves 0,1 so that every
// WMMA runs with full EXEC — required by the ISA); only waves 0,1 store y.
__global__ __launch_bounds__(128) void rnn_kernel(const float* __restrict__ x_pred,
                                                  const __bf16* __restrict__ W,
                                                  float* __restrict__ y_out,
                                                  float* __restrict__ x_final) {
  __shared__ __bf16 x_bf[16 * 64];
  __shared__ __bf16 w_bf[16 * 64];

  const int tid = threadIdx.x;
  const int lane = tid & 31;
  const int wave = tid >> 5;          // 0..3 -> N tile
  const int nb = wave * 16;
  const int bbase = blockIdx.x * 16;  // batch-row base

  // x0 = 0
  for (int i = tid; i < 16 * 64; i += 128) x_bf[i] = (__bf16)0.0f;

  // Weight B-fragments (live in registers for the whole scan)
  const __bf16* Wz_x = W + 0;
  const __bf16* Wz_u = W + 4096;
  const __bf16* Wx_x = W + 6144;
  const __bf16* Wx_u = W + 10240;
  const __bf16* Wx_w = W + 12288;
  const __bf16* Wy_x = W + 16384;
  const __bf16* Wy_u = W + 18432;
  const __bf16* Wy_w = W + 19456;

  const int yb = (wave & 1) * 16;  // y col tile (stored only by waves 0,1)

  v16bf bzx0 = load_b_frag(Wz_x,            NWD, nb, lane);
  v16bf bzx1 = load_b_frag(Wz_x + 32 * NWD, NWD, nb, lane);
  v16bf bzu  = load_b_frag(Wz_u,            NWD, nb, lane);
  v16bf bxx0 = load_b_frag(Wx_x,            NXD, nb, lane);
  v16bf bxx1 = load_b_frag(Wx_x + 32 * NXD, NXD, nb, lane);
  v16bf bxu  = load_b_frag(Wx_u,            NXD, nb, lane);
  v16bf bxw0 = load_b_frag(Wx_w,            NXD, nb, lane);
  v16bf bxw1 = load_b_frag(Wx_w + 32 * NXD, NXD, nb, lane);
  v16bf byx0 = load_b_frag(Wy_x,            NYD, yb, lane);
  v16bf byx1 = load_b_frag(Wy_x + 32 * NYD, NYD, yb, lane);
  v16bf byu  = load_b_frag(Wy_u,            NYD, yb, lane);
  v16bf byw0 = load_b_frag(Wy_w,            NYD, yb, lane);
  v16bf byw1 = load_b_frag(Wy_w + 32 * NYD, NYD, yb, lane);

  __syncthreads();

  const float* uBase = x_pred + (long)bbase * TSTEPS * NUD;  // row stride T*NU
  const long uStride = (long)TSTEPS * NUD;

  // Software pipeline: u for t=0 loaded up front.
  v16bf ua = load_a_global_f32(uBase, uStride, lane);
  v8f xlast = {};

  for (int t = 0; t < TSTEPS; ++t) {
    // A-fragments of current state (barrier at loop end made these visible)
    v16bf xa0 = load_a_lds(x_bf, 0, lane);
    v16bf xa1 = load_a_lds(x_bf, 32, lane);

    // z tile = x @ Wz_x + u @ Wz_u   (t_inv already folded in)
    v8f zacc = {};
    zacc = WMMA_BF16(xa0, bzx0, zacc);
    zacc = WMMA_BF16(xa1, bzx1, zacc);
    zacc = WMMA_BF16(ua,  bzu,  zacc);

    // Issue the w-independent partials of x_next / y before the barrier so
    // only 2 WMMAs per output remain on the post-barrier critical path.
    v8f xacc = {};
    xacc = WMMA_BF16(xa0, bxx0, xacc);
    xacc = WMMA_BF16(xa1, bxx1, xacc);
    xacc = WMMA_BF16(ua,  bxu,  xacc);

    // y partials in ALL waves (no EXEC masking around WMMA).
    v8f yacc = {};
    yacc = WMMA_BF16(xa0, byx0, yacc);
    yacc = WMMA_BF16(xa1, byx1, yacc);
    yacc = WMMA_BF16(ua,  byu,  yacc);

    // Pipeline: fetch next timestep's input now; latency overlaps tanh+barrier.
    v16bf ua_next = ua;
    if (t + 1 < TSTEPS)
      ua_next = load_a_global_f32(uBase + (long)(t + 1) * NUD, uStride, lane);
    if (t + 2 < TSTEPS) {
      const float* np = uBase + (long)(t + 2) * NUD + (long)(lane & 15) * uStride;
      __builtin_prefetch(np, 0, 1);   // global_prefetch_b8
    }

    // w = tanh(z) -> LDS (bf16, row-major [16][64]); branch-free tanh
    {
      const int col = nb + (lane & 15);
      const int rbase = (lane >> 4) * 8;
#pragma unroll
      for (int v = 0; v < 8; ++v)
        w_bf[(rbase + v) * 64 + col] = (__bf16)fast_tanh(zacc[v]);
    }
    __syncthreads();  // w visible; all waves hold old x in registers

    v16bf wa0 = load_a_lds(w_bf, 0, lane);
    v16bf wa1 = load_a_lds(w_bf, 32, lane);

    // Finish x_next: += w @ Wx_w   (Y_inv folded in)
    xacc = WMMA_BF16(wa0, bxw0, xacc);
    xacc = WMMA_BF16(wa1, bxw1, xacc);

    // Finish y: += w @ Wy_w (all waves; only waves 0,1 store)
    yacc = WMMA_BF16(wa0, byw0, yacc);
    yacc = WMMA_BF16(wa1, byw1, yacc);
    if (wave < 2) {
      const int col = yb + (lane & 15);
      const int rbase = (lane >> 4) * 8;
#pragma unroll
      for (int v = 0; v < 8; ++v) {
        long row = bbase + rbase + v;
        y_out[(row * TSTEPS + t) * NYD + col] = yacc[v];
      }
    }

    // commit x_next state (bf16) back to LDS; keep f32 copy for x_final
    {
      const int col = nb + (lane & 15);
      const int rbase = (lane >> 4) * 8;
#pragma unroll
      for (int v = 0; v < 8; ++v)
        x_bf[(rbase + v) * 64 + col] = (__bf16)xacc[v];
    }
    xlast = xacc;
    ua = ua_next;
    __syncthreads();  // new state visible before next iteration's loads
  }

  // x_final (f32) from the last step's accumulator
  {
    const int col = nb + (lane & 15);
    const int rbase = (lane >> 4) * 8;
#pragma unroll
    for (int v = 0; v < 8; ++v)
      x_final[(long)(bbase + rbase + v) * NXD + col] = xlast[v];
  }
}

extern "C" void kernel_launch(void* const* d_in, const int* in_sizes, int n_in,
                              void* d_out, int out_size, void* d_ws, size_t ws_size,
                              hipStream_t stream) {
  const float* x_pred  = (const float*)d_in[0];
  const float* Y       = (const float*)d_in[1];
  const float* lambdas = (const float*)d_in[2];
  const float* A       = (const float*)d_in[3];
  const float* B1      = (const float*)d_in[4];
  const float* B2      = (const float*)d_in[5];
  const float* C1      = (const float*)d_in[6];
  const float* D11     = (const float*)d_in[7];
  const float* D12     = (const float*)d_in[8];
  const float* C2      = (const float*)d_in[9];
  const float* D21     = (const float*)d_in[10];

  float*  Yinv = (float*)d_ws;
  __bf16* W    = (__bf16*)((char*)d_ws + 64 * 64 * sizeof(float));

  float* y_out   = (float*)d_out;
  float* x_final = y_out + (size_t)BBATCH * TSTEPS * NYD;

  prep_inverse<<<1, 128, 0, stream>>>(Y, Yinv);
  prep_weights<<<1, 256, 0, stream>>>(lambdas, A, B1, B2, C1, D11, D12, C2, D21,
                                      Yinv, W);
  rnn_kernel<<<BBATCH / 16, 128, 0, stream>>>(x_pred, W, y_out, x_final);
}
